// MLPSalmonn_36172214567205
// MI455X (gfx1250) — compile-verified, matrix operand order checked by reading, not run on hardware
//
#include <hip/hip_runtime.h>
#include <math.h>

// ---------------------------------------------------------------------------
// Problem dims (fixed by the reference)
// ---------------------------------------------------------------------------
#define S_DIM 4096
#define D_DIM 5120
#define H_DIM 5120
#define V_DIM 32000
#define P_DIM 1024
#define SCALE_F   0.2f
#define INV_TEMP  10.0f
#define LN_EPS_F  1e-5f
#define L2_EPS_F  1e-12f

typedef __attribute__((ext_vector_type(16))) __bf16 v16bf;
typedef __attribute__((ext_vector_type(8)))  float  v8f;
typedef __attribute__((ext_vector_type(4)))  int    v4i;

union FragU { v16bf v; uint4 q[2]; };

// Direct global -> LDS async staging (CDNA5 GLOBAL_LOAD_ASYNC_TO_LDS_B128),
// guarded so the file still compiles on toolchains without the builtin.
#if __has_builtin(__builtin_amdgcn_global_load_async_to_lds_b128)
#define USE_ASYNC_LDS 1
#else
#define USE_ASYNC_LDS 0
#endif

// Address-space-typed pointer casts for the async builtin (param 1 is a
// global-AS int4*, per the compiler diagnostic; LDS side cast to AS(3) int4*,
// which also implicitly converts to void* if the prototype uses that).
#define GAS_V4I(p) ((__attribute__((address_space(1))) v4i*)(p))
#define LAS_V4I(p) ((__attribute__((address_space(3))) v4i*)(p))

// ---------------------------------------------------------------------------
// Block-wide reductions (256 threads)
// ---------------------------------------------------------------------------
__device__ __forceinline__ float blockReduceSum(float v, float* sbuf) {
  const int tid = threadIdx.x;
  sbuf[tid] = v;
  __syncthreads();
  for (int s = 128; s > 0; s >>= 1) {
    if (tid < s) sbuf[tid] += sbuf[tid + s];
    __syncthreads();
  }
  float r = sbuf[0];
  __syncthreads();
  return r;
}

__device__ __forceinline__ float blockReduceMax(float v, float* sbuf) {
  const int tid = threadIdx.x;
  sbuf[tid] = v;
  __syncthreads();
  for (int s = 128; s > 0; s >>= 1) {
    if (tid < s) sbuf[tid] = fmaxf(sbuf[tid], sbuf[tid + s]);
    __syncthreads();
  }
  float r = sbuf[0];
  __syncthreads();
  return r;
}

// ---------------------------------------------------------------------------
// Generic bf16 WMMA GEMM:  C[M,N] (f32) = A[M,K] (bf16, row-major, lda=K)
//                                       x B (f32, converted to bf16 on the fly)
// TRANSB = 0:  B[k][n] = Bsrc[k*ldb + n]   (row-major weights / vocab)
// TRANSB = 1:  B[k][n] = Bsrc[n*ldb + k]   (vocab^T for the similarity GEMM)
// Block tile 128x128, BK=32; 8 waves of 32 lanes; wave tile 32x64 (2x4 WMMA).
// All of M,N divisible by 128 and K by 32 for the shapes used here.
// ---------------------------------------------------------------------------
template <bool TRANSB>
__global__ __launch_bounds__(256)
void gemm_bf16_wmma(const __bf16* __restrict__ A, const float* __restrict__ B,
                    float* __restrict__ C, int M, int N, int K, int ldb) {
  __shared__ __bf16 As[128][32];   // A tile  [m][k]
  __shared__ __bf16 Bs[128][32];   // B tile  [n][k]  (K-contiguous per column)

  const int tid    = threadIdx.x;
  const int lane   = tid & 31;
  const int wid    = tid >> 5;
  const int wave_m = wid & 3;      // 4 wave-rows  x 32 rows
  const int wave_n = wid >> 2;     // 2 wave-cols  x 64 cols
  const int lm     = lane & 15;
  const int hh     = lane >> 4;    // half selector for the 16-bit WMMA layout

  const int gm0 = blockIdx.y * 128;
  const int gn0 = blockIdx.x * 128;

  v8f acc[2][4];
  v8f zero;
#pragma unroll
  for (int e = 0; e < 8; ++e) zero[e] = 0.0f;
#pragma unroll
  for (int i = 0; i < 2; ++i)
#pragma unroll
    for (int j = 0; j < 4; ++j) acc[i][j] = zero;

  const int ar  = tid >> 1;        // 0..127  A-tile row
  const int ac0 = (tid & 1) * 16;  // 0 or 16 A-tile k-chunk

  for (int k0 = 0; k0 < K; k0 += 32) {
    // ---- stage A tile (bf16, already converted) -------------------------
    {
      const __bf16* src = A + (size_t)(gm0 + ar) * K + (k0 + ac0);
#if USE_ASYNC_LDS
      __builtin_amdgcn_global_load_async_to_lds_b128(
          GAS_V4I(src), LAS_V4I(&As[ar][ac0]), 0, 0);
      __builtin_amdgcn_global_load_async_to_lds_b128(
          GAS_V4I(src + 8), LAS_V4I(&As[ar][ac0 + 8]), 0, 0);
      if (k0 + 32 < K) __builtin_prefetch(src + 32, 0, 0);
#else
      const uint4* s4 = (const uint4*)src;
      uint4 a0 = s4[0];
      uint4 a1 = s4[1];
      if (k0 + 32 < K) __builtin_prefetch(src + 32, 0, 0);
      *(uint4*)&As[ar][ac0]     = a0;
      *(uint4*)&As[ar][ac0 + 8] = a1;
#endif
    }

    // ---- stage B tile (f32 -> bf16 conversion in-flight) ----------------
    if (!TRANSB) {
      // each thread: 4(k) x 4(n) micro-tile -> 4x ds_store_b64 (k-contiguous)
      const int kg = tid >> 5;   // 0..7  -> k = kg*4 + i
      const int ng = tid & 31;   // 0..31 -> n = ng*4 + j
      const float* src = B + (size_t)(k0 + kg * 4) * ldb + (gn0 + ng * 4);
      float4 r0 = ((const float4*)(src))[0];
      float4 r1 = ((const float4*)(src + (size_t)ldb))[0];
      float4 r2 = ((const float4*)(src + 2 * (size_t)ldb))[0];
      float4 r3 = ((const float4*)(src + 3 * (size_t)ldb))[0];
      if (k0 + 32 < K) __builtin_prefetch(src + (size_t)32 * ldb, 0, 0);
      float c0[4] = {r0.x, r0.y, r0.z, r0.w};
      float c1[4] = {r1.x, r1.y, r1.z, r1.w};
      float c2[4] = {r2.x, r2.y, r2.z, r2.w};
      float c3[4] = {r3.x, r3.y, r3.z, r3.w};
#pragma unroll
      for (int j = 0; j < 4; ++j) {
        union { __bf16 h[4]; uint2 q; } t;
        t.h[0] = (__bf16)c0[j];
        t.h[1] = (__bf16)c1[j];
        t.h[2] = (__bf16)c2[j];
        t.h[3] = (__bf16)c3[j];
        *(uint2*)&Bs[ng * 4 + j][kg * 4] = t.q;
      }
    } else {
      const int bn  = tid >> 1;        // 0..127
      const int bk0 = (tid & 1) * 16;  // 0 or 16
      const float* src = B + (size_t)(gn0 + bn) * ldb + (k0 + bk0);
      const float4* s4 = (const float4*)src;
      float4 f0 = s4[0], f1 = s4[1], f2 = s4[2], f3 = s4[3];
      if (k0 + 32 < K) __builtin_prefetch(src + 32, 0, 0);
      float fv[16] = {f0.x, f0.y, f0.z, f0.w, f1.x, f1.y, f1.z, f1.w,
                      f2.x, f2.y, f2.z, f2.w, f3.x, f3.y, f3.z, f3.w};
      union { __bf16 h[16]; uint4 q[2]; } t;
#pragma unroll
      for (int j = 0; j < 16; ++j) t.h[j] = (__bf16)fv[j];
      *(uint4*)&Bs[bn][bk0]     = t.q[0];
      *(uint4*)&Bs[bn][bk0 + 8] = t.q[1];
    }

#if USE_ASYNC_LDS
#if __has_builtin(__builtin_amdgcn_s_wait_asynccnt)
    __builtin_amdgcn_s_wait_asynccnt(0);
#else
    asm volatile("s_wait_asynccnt 0x0" ::: "memory");
#endif
#endif
    __syncthreads();

    // ---- load fragments per the CDNA5 16-bit WMMA VGPR layout -----------
    // lane (hh=lane>>4): element e -> k = (e<8?0:16) + hh*8 + (e&7)
    // -> two contiguous 16B chunks at k-offsets hh*8 and 16+hh*8.
    FragU afrag[2], bfrag[4];
#pragma unroll
    for (int mt = 0; mt < 2; ++mt) {
      const __bf16* p = &As[wave_m * 32 + mt * 16 + lm][0];
      afrag[mt].q[0] = *(const uint4*)(p + hh * 8);
      afrag[mt].q[1] = *(const uint4*)(p + 16 + hh * 8);
    }
#pragma unroll
    for (int nt = 0; nt < 4; ++nt) {
      const __bf16* p = &Bs[wave_n * 64 + nt * 16 + lm][0];
      bfrag[nt].q[0] = *(const uint4*)(p + hh * 8);
      bfrag[nt].q[1] = *(const uint4*)(p + 16 + hh * 8);
    }

#pragma unroll
    for (int mt = 0; mt < 2; ++mt)
#pragma unroll
      for (int nt = 0; nt < 4; ++nt)
        acc[mt][nt] = __builtin_amdgcn_wmma_f32_16x16x32_bf16(
            false, afrag[mt].v, false, bfrag[nt].v,
            (short)0, acc[mt][nt], false, false);
    __syncthreads();
  }

  // ---- writeback: C/D layout -> VGPR r holds row hh*8+r, col = lane&15 ---
#pragma unroll
  for (int mt = 0; mt < 2; ++mt) {
    const int row0 = gm0 + wave_m * 32 + mt * 16 + hh * 8;
#pragma unroll
    for (int nt = 0; nt < 4; ++nt) {
      const int col = gn0 + wave_n * 64 + nt * 16 + lm;
#pragma unroll
      for (int r = 0; r < 8; ++r)
        C[(size_t)(row0 + r) * N + col] = acc[mt][nt][r];
    }
  }
}

// ---------------------------------------------------------------------------
// Gather label rows and convert to bf16:  x_bf[p][d] = bf16(emb[pos[p]][d])
// ---------------------------------------------------------------------------
__global__ void gather_cvt_kernel(const float* __restrict__ emb,
                                  const int* __restrict__ pos,
                                  __bf16* __restrict__ xb) {
  const int idx = blockIdx.x * blockDim.x + threadIdx.x;  // < P*D
  const int p = idx / D_DIM;
  const int d = idx - p * D_DIM;
  xb[idx] = (__bf16)emb[(size_t)pos[p] * D_DIM + d];
}

// ---------------------------------------------------------------------------
// Per-row LayerNorm (+b1) then exact-erf GELU, emit bf16 activations.
// ---------------------------------------------------------------------------
__global__ __launch_bounds__(256)
void ln_gelu_kernel(const float* __restrict__ h, const float* __restrict__ b1,
                    const float* __restrict__ g, const float* __restrict__ b,
                    __bf16* __restrict__ act) {
  __shared__ float sbuf[256];
  const int p = blockIdx.x;
  const float* hr = h + (size_t)p * H_DIM;
  float s = 0.f, ss = 0.f;
  for (int d = threadIdx.x; d < H_DIM; d += 256) {
    float v = hr[d] + b1[d];
    s += v;
    ss += v * v;
  }
  const float tot  = blockReduceSum(s, sbuf);
  const float tot2 = blockReduceSum(ss, sbuf);
  const float mean = tot * (1.0f / H_DIM);
  const float var  = tot2 * (1.0f / H_DIM) - mean * mean;
  const float rstd = rsqrtf(var + LN_EPS_F);
  __bf16* ar = act + (size_t)p * H_DIM;
  for (int d = threadIdx.x; d < H_DIM; d += 256) {
    float v  = (hr[d] + b1[d] - mean) * rstd * g[d] + b[d];
    float ge = 0.5f * v * (1.0f + erff(v * 0.70710678118654752f));
    ar[d] = (__bf16)ge;
  }
}

// ---------------------------------------------------------------------------
// transformed = x + SCALE*(y + b2); row-l2-normalize; emit bf16.
// ---------------------------------------------------------------------------
__global__ __launch_bounds__(256)
void residual_l2_kernel(const float* __restrict__ emb,
                        const int* __restrict__ pos,
                        const float* __restrict__ y,
                        const float* __restrict__ b2,
                        __bf16* __restrict__ tn) {
  __shared__ float sbuf[256];
  __shared__ float trow[D_DIM];  // 20 KB row buffer in LDS
  const int p = blockIdx.x;
  const float* er = emb + (size_t)pos[p] * D_DIM;
  const float* yr = y + (size_t)p * D_DIM;
  float ss = 0.f;
  for (int d = threadIdx.x; d < D_DIM; d += 256) {
    float t = er[d] + SCALE_F * (yr[d] + b2[d]);
    trow[d] = t;
    ss += t * t;
  }
  const float tot = blockReduceSum(ss, sbuf);
  const float inv = 1.0f / fmaxf(sqrtf(tot), L2_EPS_F);
  __bf16* tr = tn + (size_t)p * D_DIM;
  for (int d = threadIdx.x; d < D_DIM; d += 256)
    tr[d] = (__bf16)(trow[d] * inv);
}

// ---------------------------------------------------------------------------
// inv_norm[v] = 1 / max(||vocab[v]||, eps)
// ---------------------------------------------------------------------------
__global__ __launch_bounds__(256)
void vnorm_kernel(const float* __restrict__ vocab,
                  float* __restrict__ inv_norm) {
  __shared__ float sbuf[256];
  const int v = blockIdx.x;
  const float* r = vocab + (size_t)v * D_DIM;
  float ss = 0.f;
  for (int d = threadIdx.x; d < D_DIM; d += 256) {
    float x = r[d];
    ss += x * x;
  }
  const float tot = blockReduceSum(ss, sbuf);
  if (threadIdx.x == 0) inv_norm[v] = 1.0f / fmaxf(sqrtf(tot), L2_EPS_F);
}

// ---------------------------------------------------------------------------
// Per-row softmax over V of (sim * inv_norm[v] / TEMP); emit bf16 weights.
// 3 passes over the row; sim fits in the 192 MB L2 so re-reads are cheap.
// ---------------------------------------------------------------------------
__global__ __launch_bounds__(256)
void softmax_cvt_kernel(const float* __restrict__ sim,
                        const float* __restrict__ inv_norm,
                        __bf16* __restrict__ w) {
  __shared__ float sbuf[256];
  const int p = blockIdx.x;
  const float* sr = sim + (size_t)p * V_DIM;
  float mx = -3.4e38f;
  for (int v = threadIdx.x; v < V_DIM; v += 256)
    mx = fmaxf(mx, sr[v] * inv_norm[v] * INV_TEMP);
  mx = blockReduceMax(mx, sbuf);
  float sum = 0.f;
  for (int v = threadIdx.x; v < V_DIM; v += 256)
    sum += expf(sr[v] * inv_norm[v] * INV_TEMP - mx);
  sum = blockReduceSum(sum, sbuf);
  const float isum = 1.0f / sum;
  __bf16* wr = w + (size_t)p * V_DIM;
  for (int v = threadIdx.x; v < V_DIM; v += 256)
    wr[v] = (__bf16)(expf(sr[v] * inv_norm[v] * INV_TEMP - mx) * isum);
}

// ---------------------------------------------------------------------------
// out[pos[p]][d] = final[p][d]
// ---------------------------------------------------------------------------
__global__ void scatter_kernel(const float* __restrict__ fin,
                               const int* __restrict__ pos,
                               float* __restrict__ out) {
  const int idx = blockIdx.x * blockDim.x + threadIdx.x;  // < P*D
  const int p = idx / D_DIM;
  const int d = idx - p * D_DIM;
  out[(size_t)pos[p] * D_DIM + d] = fin[idx];
}

// ---------------------------------------------------------------------------
extern "C" void kernel_launch(void* const* d_in, const int* in_sizes, int n_in,
                              void* d_out, int out_size, void* d_ws,
                              size_t ws_size, hipStream_t stream) {
  const float* emb   = (const float*)d_in[0];
  const int*   pos   = (const int*)d_in[1];
  const float* vocab = (const float*)d_in[2];
  const float* W1    = (const float*)d_in[3];
  const float* b1    = (const float*)d_in[4];
  const float* ln_g  = (const float*)d_in[5];
  const float* ln_b  = (const float*)d_in[6];
  const float* W2    = (const float*)d_in[7];
  const float* b2    = (const float*)d_in[8];
  float* out = (float*)d_out;

  // workspace carve (256B aligned), ~291 MB total
  char* ws = (char*)d_ws;
  size_t off = 0;
  auto alloc = [&](size_t bytes) -> void* {
    void* p = ws + off;
    off += (bytes + 255) & ~(size_t)255;
    return p;
  };
  __bf16* x_bf   = (__bf16*)alloc((size_t)P_DIM * D_DIM * 2);
  float*  h      = (float*)alloc((size_t)P_DIM * H_DIM * 4);
  __bf16* act_bf = (__bf16*)alloc((size_t)P_DIM * H_DIM * 2);
  float*  y      = (float*)alloc((size_t)P_DIM * D_DIM * 4);
  __bf16* tn_bf  = (__bf16*)alloc((size_t)P_DIM * D_DIM * 2);
  float*  invn   = (float*)alloc((size_t)V_DIM * 4);
  float*  sim    = (float*)alloc((size_t)P_DIM * V_DIM * 4);
  __bf16* w_bf   = (__bf16*)alloc((size_t)P_DIM * V_DIM * 2);
  float*  fin    = (float*)alloc((size_t)P_DIM * D_DIM * 4);

  // out starts as a copy of embeddings; label rows overwritten at the end
  (void)hipMemcpyAsync(out, emb, (size_t)S_DIM * D_DIM * sizeof(float),
                       hipMemcpyDeviceToDevice, stream);

  gather_cvt_kernel<<<(P_DIM * D_DIM) / 256, 256, 0, stream>>>(emb, pos, x_bf);
  vnorm_kernel<<<V_DIM, 256, 0, stream>>>(vocab, invn);

  // G1: h = x @ W1            [1024,5120] x [5120,5120]
  gemm_bf16_wmma<false><<<dim3(H_DIM / 128, P_DIM / 128), 256, 0, stream>>>(
      x_bf, W1, h, P_DIM, H_DIM, D_DIM, H_DIM);
  ln_gelu_kernel<<<P_DIM, 256, 0, stream>>>(h, b1, ln_g, ln_b, act_bf);

  // G2: y = act @ W2          [1024,5120] x [5120,5120]
  gemm_bf16_wmma<false><<<dim3(D_DIM / 128, P_DIM / 128), 256, 0, stream>>>(
      act_bf, W2, y, P_DIM, D_DIM, H_DIM, D_DIM);
  residual_l2_kernel<<<P_DIM, 256, 0, stream>>>(emb, pos, y, b2, tn_bf);

  // G3: sim = tn @ vocab^T    [1024,5120] x [5120,32000]  (B transposed)
  gemm_bf16_wmma<true><<<dim3(V_DIM / 128, P_DIM / 128), 256, 0, stream>>>(
      tn_bf, vocab, sim, P_DIM, V_DIM, D_DIM, D_DIM);
  softmax_cvt_kernel<<<P_DIM, 256, 0, stream>>>(sim, invn, w_bf);

  // G4: fin = w @ vocab       [1024,32000] x [32000,5120]
  gemm_bf16_wmma<false><<<dim3(D_DIM / 128, P_DIM / 128), 256, 0, stream>>>(
      w_bf, vocab, fin, P_DIM, D_DIM, V_DIM, D_DIM);
  scatter_kernel<<<(P_DIM * D_DIM) / 256, 256, 0, stream>>>(fin, pos, out);

  (void)in_sizes; (void)n_in; (void)out_size; (void)ws_size;
}
